// Graph_ConvNet_LeNet5_66460323938416
// MI455X (gfx1250) — matrix operand order repaired and checked by reading, not compiled, not run on hardware
//
#include <hip/hip_runtime.h>
#include <hip/hip_bf16.h>

typedef _Float16 half_t;
typedef __attribute__((ext_vector_type(16))) _Float16 v16h;
typedef __attribute__((ext_vector_type(8)))  float    v8f;

#define B_    64
#define V1_   4096
#define V2_   1024
#define DF_   8
#define KCH   25
#define F1_   32
#define F2_   64
#define FC1F_ 512
#define FC2F_ 10
#define E1_   65536
#define E2_   16384
#define C1_   512      // DF_*B_
#define C2_   2048     // F1_*B_
#define FC1IN 16384

static constexpr size_t SLOT_F  = (size_t)V1_ * C1_;        // 2097152 floats (== V2_*C2_)
static constexpr size_t STACK_B = (size_t)KCH * SLOT_F * 4; // 209,715,200 B
static constexpr size_t OFF_OUT1 = STACK_B;                         // 262144*32*4
static constexpr size_t OFF_OUT2 = OFF_OUT1 + (size_t)33554432;     // 65536*64*4
static constexpr size_t OFF_FCIN = OFF_OUT2 + (size_t)16777216;     // 64*16384*2
static constexpr size_t OFF_W1   = OFF_FCIN + (size_t)2097152;      // 224*32*2 (pad)
static constexpr size_t OFF_W2   = OFF_W1   + (size_t)16384;        // 800*64*2
static constexpr size_t OFF_W3   = OFF_W2   + (size_t)102400;       // 16384*512*2
static constexpr size_t OFF_FC1O = OFF_W3   + (size_t)16777216;     // 64*512*4

__device__ __forceinline__ void atomAddF(float* p, float v) {
  __hip_atomic_fetch_add(p, v, __ATOMIC_RELAXED, __HIP_MEMORY_SCOPE_AGENT);
}

// ---- x[b][v][f] -> stack slot0 [v][b][f] --------------------------------
__global__ void k_pack_x0(const float* __restrict__ x, float* __restrict__ s0) {
  unsigned i = blockIdx.x * 256u + threadIdx.x;            // over B*V1*DF = 2^21
  if (i >= (unsigned)(B_ * V1_ * DF_)) return;
  unsigned f = i & (DF_ - 1);
  unsigned v = (i >> 3) & (V1_ - 1);
  unsigned b = i >> 15;
  s0[((size_t)v * B_ + b) * DF_ + f] = x[i];
}

__global__ void k_zero(float* __restrict__ p, unsigned n) {
  unsigned i = blockIdx.x * 256u + threadIdx.x;
  if (i < n) p[i] = 0.0f;
}

__global__ void k_negcopy(const float* __restrict__ s, float* __restrict__ d, unsigned n) {
  unsigned i = blockIdx.x * 256u + threadIdx.x;
  if (i < n) d[i] = -s[i];
}

// ---- COO SPMM with fp32 global atomics: yout[r] += scale*val*yin[c] -----
__global__ void k_spmm(const int* __restrict__ rows, const int* __restrict__ cols,
                       const float* __restrict__ vals, const float* __restrict__ yin,
                       float* __restrict__ yout, int E, int cshift, int C, float scale) {
  unsigned gid = blockIdx.x * 256u + threadIdx.x;
  unsigned e = gid >> cshift;
  if (e >= (unsigned)E) return;
  unsigned cb = (gid & ((1u << cshift) - 1u)) << 2;
  int r = rows[e], c = cols[e];
  float v = vals[e] * scale;
  const float4 y = *reinterpret_cast<const float4*>(yin + (size_t)c * C + cb);
  float* o = yout + (size_t)r * C + cb;
  atomAddF(o + 0, v * y.x);
  atomAddF(o + 1, v * y.y);
  atomAddF(o + 2, v * y.z);
  atomAddF(o + 3, v * y.w);
}

// ---- W [N, Kdim] fp32 -> Wp [Kpad, N] f16 (zero-padded rows) ------------
__global__ void k_pack_w(const float* __restrict__ W, half_t* __restrict__ Wp,
                         int nShift, int Kdim, int total) {
  int i = blockIdx.x * 256 + threadIdx.x;            // i = ck*N + n
  if (i >= total) return;
  int N = 1 << nShift;
  int n = i & (N - 1);
  int ck = i >> nShift;
  Wp[i] = (ck < Kdim) ? (half_t)W[(size_t)n * Kdim + ck] : (half_t)0.0f;
}

// ---- Chebyshev Theta GEMM: out[b*V+v, o] = relu(A @ Wp + bias) ----------
// A[m][ck]: ck = f*25+kc ; A = stack[kc][v][b][f], stack layout [k][V][64][Fin]
// Fully compile-time shape so acc[] stays in fixed VGPRs (no v_movrel).
template <int V, int VSHIFT, int FIN, int FSHIFT, int KDIM, int NSTEPS, int N>
__global__ __launch_bounds__(256)
void k_gemm_cheb(const float* __restrict__ stack, const half_t* __restrict__ Wp,
                 const float* __restrict__ bias, float* __restrict__ out) {
  constexpr int LDA = 40, LDB = 72;
  constexpr int NB = N / 16;        // 2 or 4
  __shared__ half_t sA[128 * LDA];
  __shared__ half_t sB[32 * LDB];
  const int tid = threadIdx.x;
  const int lane = tid & 31, wave = tid >> 5;
  const int m0 = blockIdx.x * 128;
  v8f acc[NB];
#pragma unroll
  for (int i = 0; i < NB; ++i)
#pragma unroll
    for (int h = 0; h < 8; ++h) acc[i][h] = 0.0f;

  for (int kt = 0; kt < NSTEPS; ++kt) {
    const int ck0 = kt * 32;
    // stage A tile (128x32) with (f,kc,b,v) gather + f32->f16 convert
#pragma unroll
    for (int t0 = 0; t0 < 128 * 32; t0 += 256) {
      int t = t0 + tid;
      int r = t >> 5, cc = t & 31;
      int ck = ck0 + cc;
      int m = m0 + r;
      int b = m >> VSHIFT, v = m & (V - 1);
      float val = 0.0f;
      if (ck < KDIM) {
        int f = ck / KCH;
        int kc = ck - f * KCH;
        val = stack[(((size_t)kc * V + v) * B_ + b) * FIN + f];
      }
      sA[r * LDA + cc] = (half_t)val;
    }
    // stage B tile (32xN) from pre-packed f16 weights
#pragma unroll
    for (int t0 = 0; t0 < 32 * N; t0 += 256) {
      int t = t0 + tid;
      int kk = t / N, n = t & (N - 1);
      sB[kk * LDB + n] = Wp[(size_t)(ck0 + kk) * N + n];
    }
    __syncthreads();
    // A fragment: lane<16 holds K 0..7,16..23 ; lane>=16 holds K 8..15,24..31
    v16h a;
    {
      int mrow = wave * 16 + (lane & 15);
      int khalf = lane >> 4;
#pragma unroll
      for (int h = 0; h < 16; ++h) {
        int K = ((h < 8) ? 0 : 16) + khalf * 8 + (h & 7);
        a[h] = sA[mrow * LDA + K];
      }
    }
#pragma unroll
    for (int nb = 0; nb < NB; ++nb) {
      v16h bf;
#pragma unroll
      for (int h = 0; h < 16; ++h) bf[h] = sB[(((h < 8) ? 0 : 16) + (lane >> 4) * 8 + (h & 7)) * LDB + nb * 16 + (lane & 15)];
      acc[nb] = __builtin_amdgcn_wmma_f32_16x16x32_f16(
          false, a, false, bf, (short)0, acc[nb], false, false);
    }
    __syncthreads();
  }
  // epilogue: D lane mapping n=lane&15, rows = (lane>>4)*8 + i ; fused bias+ReLU
#pragma unroll
  for (int nb = 0; nb < NB; ++nb) {
    int n = nb * 16 + (lane & 15);
    float bn = bias[n];
    int mrow = m0 + wave * 16 + (lane >> 4) * 8;
#pragma unroll
    for (int i = 0; i < 8; ++i) {
      float vv = acc[nb][i] + bn;
      out[(size_t)(mrow + i) * N + n] = vv > 0.0f ? vv : 0.0f;
    }
  }
}

// ---- relu'd layer1 out [b*V1+v][32] -> maxpool/4 -> stack2 slot0 [vp][b][f]
__global__ void k_pool1(const float* __restrict__ h, float* __restrict__ s0) {
  unsigned i = blockIdx.x * 256u + threadIdx.x;   // over V2*B*F1 = 2^21
  if (i >= (unsigned)(V2_ * B_ * F1_)) return;
  unsigned f = i & 31, b = (i >> 5) & 63, vp = i >> 11;
  size_t base = ((size_t)b * V1_ + 4 * vp) * F1_ + f;
  float m = fmaxf(fmaxf(h[base], h[base + F1_]), fmaxf(h[base + 2 * F1_], h[base + 3 * F1_]));
  s0[i] = m;   // i == (vp*64 + b)*32 + f
}

// ---- layer2 out [b*V2+v][64] -> maxpool/4 -> fcin f16 [b][vq*64+f] ------
__global__ void k_pool2(const float* __restrict__ h, half_t* __restrict__ fcin) {
  unsigned i = blockIdx.x * 256u + threadIdx.x;   // over B*256*F2 = 2^20
  if (i >= (unsigned)(B_ * 256 * F2_)) return;
  unsigned f = i & 63, vq = (i >> 6) & 255, b = i >> 14;
  size_t base = ((size_t)b * V2_ + 4 * vq) * F2_ + f;
  float m = fmaxf(fmaxf(h[base], h[base + F2_]), fmaxf(h[base + 2 * F2_], h[base + 3 * F2_]));
  fcin[i] = (half_t)m;   // i == b*16384 + vq*64 + f
}

// ---- FC1: [64,16384] @ Wp3[16384,512] + bias, ReLU ----------------------
__global__ __launch_bounds__(256)
void k_gemm_fc1(const half_t* __restrict__ A, const half_t* __restrict__ Wp,
                const float* __restrict__ bias, float* __restrict__ out) {
  constexpr int LDA = 40, LDB = 72;
  __shared__ half_t sA[64 * LDA];
  __shared__ half_t sB[32 * LDB];
  const int tid = threadIdx.x, lane = tid & 31, wave = tid >> 5;
  const int n0 = blockIdx.x * 64;
  const int mb = wave >> 1;            // 0..3
  const int nbB = (wave & 1) * 2;      // 0 or 2
  v8f acc[2];
#pragma unroll
  for (int i = 0; i < 2; ++i)
#pragma unroll
    for (int h = 0; h < 8; ++h) acc[i][h] = 0.0f;

  for (int kt = 0; kt < FC1IN / 32; ++kt) {
    const int k0 = kt * 32;
#pragma unroll
    for (int t0 = 0; t0 < 64 * 32; t0 += 256) {
      int t = t0 + tid;
      int r = t >> 5, cc = t & 31;
      sA[r * LDA + cc] = A[(size_t)r * FC1IN + k0 + cc];
    }
#pragma unroll
    for (int t0 = 0; t0 < 32 * 64; t0 += 256) {
      int t = t0 + tid;
      int kk = t >> 6, n = t & 63;
      sB[kk * LDB + n] = Wp[(size_t)(k0 + kk) * FC1F_ + n0 + n];
    }
    __syncthreads();
    v16h a;
    {
      int mrow = mb * 16 + (lane & 15);
      int khalf = lane >> 4;
#pragma unroll
      for (int h = 0; h < 16; ++h) {
        int K = ((h < 8) ? 0 : 16) + khalf * 8 + (h & 7);
        a[h] = sA[mrow * LDA + K];
      }
    }
#pragma unroll
    for (int j = 0; j < 2; ++j) {
      v16h bf;
#pragma unroll
      for (int h = 0; h < 16; ++h) bf[h] = sB[(((h < 8) ? 0 : 16) + (lane >> 4) * 8 + (h & 7)) * LDB + (nbB + j) * 16 + (lane & 15)];
      acc[j] = __builtin_amdgcn_wmma_f32_16x16x32_f16(
          false, a, false, bf, (short)0, acc[j], false, false);
    }
    __syncthreads();
  }
#pragma unroll
  for (int j = 0; j < 2; ++j) {
    int n = n0 + (nbB + j) * 16 + (lane & 15);
    float bn = bias[n];
    int mrow = mb * 16 + (lane >> 4) * 8;
#pragma unroll
    for (int i = 0; i < 8; ++i) {
      float vv = acc[j][i] + bn;
      out[(size_t)(mrow + i) * FC1F_ + n] = vv > 0.0f ? vv : 0.0f;
    }
  }
}

// ---- FC2 (tiny): out[b][o] = h[b] . W[o] + bias[o] ----------------------
__global__ void k_fc2(const float* __restrict__ h, const float* __restrict__ W,
                      const float* __restrict__ bias, float* __restrict__ out) {
  int gid = blockIdx.x * 64 + threadIdx.x;
  if (gid >= B_ * FC2F_) return;
  int b = gid / FC2F_, o = gid - b * FC2F_;
  float s = bias[o];
  const float* hp = h + (size_t)b * FC1F_;
  const float* wp = W + (size_t)o * FC1F_;
  for (int j = 0; j < FC1F_; ++j) s += hp[j] * wp[j];
  out[gid] = s;
}

extern "C" void kernel_launch(void* const* d_in, const int* in_sizes, int n_in,
                              void* d_out, int out_size, void* d_ws, size_t ws_size,
                              hipStream_t stream) {
  const float* x     = (const float*)d_in[0];
  const int*   L1r   = (const int*)d_in[1];
  const int*   L1c   = (const int*)d_in[2];
  const float* L1v   = (const float*)d_in[3];
  const int*   L2r   = (const int*)d_in[4];
  const int*   L2c   = (const int*)d_in[5];
  const float* L2v   = (const float*)d_in[6];
  const float* cl1w  = (const float*)d_in[7];
  const float* cl1b  = (const float*)d_in[8];
  const float* cl2w  = (const float*)d_in[9];
  const float* cl2b  = (const float*)d_in[10];
  const float* fc1w  = (const float*)d_in[11];
  const float* fc1b  = (const float*)d_in[12];
  const float* fc2w  = (const float*)d_in[13];
  const float* fc2b  = (const float*)d_in[14];

  char* wsb = (char*)d_ws;
  float*  stack  = (float*)(wsb);
  float*  out1   = (float*)(wsb + OFF_OUT1);
  float*  out2   = (float*)(wsb + OFF_OUT2);
  half_t* fcin   = (half_t*)(wsb + OFF_FCIN);
  half_t* Wp1    = (half_t*)(wsb + OFF_W1);
  half_t* Wp2    = (half_t*)(wsb + OFF_W2);
  half_t* Wp3    = (half_t*)(wsb + OFF_W3);
  float*  fc1out = (float*)(wsb + OFF_FC1O);

  const unsigned slotN = (unsigned)SLOT_F;          // 2^21
  const int slotBlk = (int)(SLOT_F / 256);          // 8192

  // ---- layer 1 Chebyshev stack ----
  k_pack_x0<<<slotBlk, 256, 0, stream>>>(x, stack);
  k_zero<<<slotBlk, 256, 0, stream>>>(stack + SLOT_F, slotN);
  k_spmm<<<(E1_ << 7) / 256, 256, 0, stream>>>(L1r, L1c, L1v, stack, stack + SLOT_F,
                                               E1_, 7, C1_, 1.0f);
  for (int k = 2; k < KCH; ++k) {
    k_negcopy<<<slotBlk, 256, 0, stream>>>(stack + (size_t)(k - 2) * SLOT_F,
                                           stack + (size_t)k * SLOT_F, slotN);
    k_spmm<<<(E1_ << 7) / 256, 256, 0, stream>>>(L1r, L1c, L1v,
                                                 stack + (size_t)(k - 1) * SLOT_F,
                                                 stack + (size_t)k * SLOT_F,
                                                 E1_, 7, C1_, 2.0f);
  }
  // ---- Theta1 GEMM (M=262144, K=200->224, N=32), fused ReLU ----
  k_pack_w<<<(224 * 32) / 256, 256, 0, stream>>>(cl1w, Wp1, 5, 200, 224 * 32);
  k_gemm_cheb<V1_, 12, DF_, 3, 200, 7, 32>
      <<<(B_ * V1_) / 128, 256, 0, stream>>>(stack, Wp1, cl1b, out1);
  // ---- pool/4 -> layer 2 stack slot0 ----
  k_pool1<<<slotBlk, 256, 0, stream>>>(out1, stack);

  // ---- layer 2 Chebyshev stack (reuses same region; same slot size) ----
  k_zero<<<slotBlk, 256, 0, stream>>>(stack + SLOT_F, slotN);
  k_spmm<<<(E2_ << 9) / 256, 256, 0, stream>>>(L2r, L2c, L2v, stack, stack + SLOT_F,
                                               E2_, 9, C2_, 1.0f);
  for (int k = 2; k < KCH; ++k) {
    k_negcopy<<<slotBlk, 256, 0, stream>>>(stack + (size_t)(k - 2) * SLOT_F,
                                           stack + (size_t)k * SLOT_F, slotN);
    k_spmm<<<(E2_ << 9) / 256, 256, 0, stream>>>(L2r, L2c, L2v,
                                                 stack + (size_t)(k - 1) * SLOT_F,
                                                 stack + (size_t)k * SLOT_F,
                                                 E2_, 9, C2_, 2.0f);
  }
  // ---- Theta2 GEMM (M=65536, K=800, N=64), fused ReLU ----
  k_pack_w<<<(800 * 64) / 256, 256, 0, stream>>>(cl2w, Wp2, 6, 800, 800 * 64);
  k_gemm_cheb<V2_, 10, F1_, 5, 800, 25, 64>
      <<<(B_ * V2_) / 128, 256, 0, stream>>>(stack, Wp2, cl2b, out2);
  // ---- pool/4 -> FC input (f16) ----
  k_pool2<<<(B_ * 256 * F2_) / 256, 256, 0, stream>>>(out2, fcin);

  // ---- FC1 (M=64, K=16384, N=512), fused ReLU ----
  k_pack_w<<<(FC1IN * FC1F_) / 256, 256, 0, stream>>>(fc1w, Wp3, 9, FC1IN, FC1IN * FC1F_);
  k_gemm_fc1<<<FC1F_ / 64, 256, 0, stream>>>(fcin, Wp3, fc1b, fc1out);

  // ---- FC2 -> d_out ----
  k_fc2<<<(B_ * FC2F_ + 63) / 64, 64, 0, stream>>>(fc1out, fc2w, fc2b, (float*)d_out);
}
